// TrajPreLocalAttnLong_9397388443885
// MI455X (gfx1250) — compile-verified, matrix operand order checked by reading, not run on hardware
//
#include <hip/hip_runtime.h>
#include <hip/hip_bf16.h>
#include <cmath>
#include <cstdint>

typedef __bf16 bf16_t;
typedef __attribute__((ext_vector_type(16))) __bf16 v16bf;
typedef __attribute__((ext_vector_type(8)))  __bf16 v8bf;
typedef __attribute__((ext_vector_type(8)))  float  v8f;

#define SEQ   4096
#define TGT   1024
#define ENC_T 3072
#define HID   1024
#define INP   1152
#define H3    3072
#define NLOC  50000

// ---------------------------------------------------------------------------
// CDNA5 async global->LDS copy (ASYNCcnt path, CDNA5 ISA).
// VDST = per-lane LDS byte address, VADDR = per-lane 64-bit global address.
// Generic pointers to LDS carry the LDS offset in their low 32 bits.
// ---------------------------------------------------------------------------
__device__ __forceinline__ void async_copy_b128(void* lds_dst, const void* g_src) {
  asm volatile("global_load_async_to_lds_b128 %0, %1, off"
               :: "v"((unsigned)(uintptr_t)lds_dst), "v"(g_src)
               : "memory");
}
__device__ __forceinline__ void wait_async_le0() {
  asm volatile("s_wait_asynccnt 0x0" ::: "memory");
}
__device__ __forceinline__ void wait_async_le2() {
  asm volatile("s_wait_asynccnt 0x2" ::: "memory");
}
__device__ __forceinline__ void wait_async_le4() {
  asm volatile("s_wait_asynccnt 0x4" ::: "memory");
}

// ---------------------------------------------------------------------------
// WMMA fragment load: 16 bf16 per lane = two contiguous 8-element K runs
// (per ISA 7.12.2: e=0..7 -> K = kb..kb+7 ; e=8..15 -> K = kb+16..kb+23)
// ---------------------------------------------------------------------------
__device__ inline v16bf ld_frag(const bf16_t* p) {
  v8bf lo = *(const v8bf*)p;
  v8bf hi = *(const v8bf*)(p + 16);
  return __builtin_shufflevector(lo, hi, 0,1,2,3,4,5,6,7,8,9,10,11,12,13,14,15);
}

// ---------------------------------------------------------------------------
// Generic NT GEMM:  C[M,N] (f32) = A[M,K](bf16, row-major) . B[N,K]^T (+bias)
// B may be f32 (converted to bf16 while staging to LDS) or bf16 (async path).
// Requires: M % 64 == 0, K % 64 == 0 (true for all uses).
// N guard only exercised by the f32-B FC GEMM (bf16-B callers have N%64==0,
// which keeps per-thread async-op counts exact for the pipelined wait).
// Block tile 64x64, 8 waves (2 M x 4 N), each wave two 16x16 WMMA tiles in M.
// Double-buffered LDS; async copies for tile k+1 overlap WMMA on tile k.
// grid.x = M tiles (fastest) so blocks sharing a B stripe are L2-coresident.
// ---------------------------------------------------------------------------
template <bool B_IS_F32>
__global__ __launch_bounds__(256)
void k_gemm_nt(const bf16_t* __restrict__ A, const void* __restrict__ Bv,
               const float* __restrict__ bias, float* __restrict__ C,
               int M, int N, int K, int lda, int ldb, int ldc)
{
  constexpr int BM = 64, BN = 64, BK = 64, LDW = BK + 8;
  __shared__ bf16_t sA[2][BM * LDW];
  __shared__ bf16_t sB[2][BN * LDW];

  const int tid  = threadIdx.x;
  const int m0   = blockIdx.x * BM;
  const int n0   = blockIdx.y * BN;
  const int lane = tid & 31;
  const int wid  = tid >> 5;
  const int wm   = wid >> 2;      // 0..1
  const int wn   = wid & 3;       // 0..3
  const int ksel = lane >> 4;     // K-half select per ISA A/B layout
  const int mrow = lane & 15;     // M row (A) / N col (B) within 16-tile

  const float*  Bf = (const float*)Bv;
  const bf16_t* Bb = (const bf16_t*)Bv;

  v8f acc0 = {}; v8f acc1 = {};

  // stage one 64x64 A tile + 64x64 B tile into LDS buffer `buf`
  auto stage = [&](int k0, int buf) {
    #pragma unroll
    for (int it = 0; it < 2; ++it) {
      const int c  = tid + it * 256;         // 0..511
      const int r  = c >> 3;                 // 0..63
      const int cc = (c & 7) * 8;            // 0,8,..,56
      // A: always in-bounds, always async (2 ops/thread/tile)
      async_copy_b128(&sA[buf][r * LDW + cc],
                      A + (size_t)(m0 + r) * lda + k0 + cc);
      const int nrow = n0 + r;
      if (B_IS_F32) {
        bf16_t* d = &sB[buf][r * LDW + cc];
        if (nrow < N) {
          const float4 b0 = *(const float4*)(Bf + (size_t)nrow * ldb + k0 + cc);
          const float4 b1 = *(const float4*)(Bf + (size_t)nrow * ldb + k0 + cc + 4);
          d[0]=(bf16_t)b0.x; d[1]=(bf16_t)b0.y; d[2]=(bf16_t)b0.z; d[3]=(bf16_t)b0.w;
          d[4]=(bf16_t)b1.x; d[5]=(bf16_t)b1.y; d[6]=(bf16_t)b1.z; d[7]=(bf16_t)b1.w;
        } else {
          #pragma unroll
          for (int q = 0; q < 8; ++q) d[q] = (bf16_t)0.0f;
        }
      } else {
        // bf16 B: async (callers guarantee nrow < N; +2 ops/thread/tile)
        async_copy_b128(&sB[buf][r * LDW + cc],
                        Bb + (size_t)nrow * ldb + k0 + cc);
      }
    }
  };

  stage(0, 0);
  int buf = 0;

  for (int k0 = 0; k0 < K; k0 += BK) {
    const bool more = (k0 + BK) < K;
    if (more) {
      stage(k0 + BK, buf ^ 1);     // overlap next tile with current compute
      if (B_IS_F32) wait_async_le2(); else wait_async_le4();
    } else {
      wait_async_le0();
    }
    __syncthreads();

    // ---- two 16x16x32 WMMA steps over this 64-K tile ----
    #pragma unroll
    for (int kk = 0; kk < BK; kk += 32) {
      const int kb = kk + ksel * 8;
      const v16bf bfr = ld_frag(&sB[buf][(wn * 16 + mrow) * LDW + kb]);
      {
        const v16bf af = ld_frag(&sA[buf][(wm * 32 + mrow) * LDW + kb]);
        acc0 = __builtin_amdgcn_wmma_f32_16x16x32_bf16(
            false, af, false, bfr, (short)0, acc0, false, false);
      }
      {
        const v16bf af = ld_frag(&sA[buf][(wm * 32 + 16 + mrow) * LDW + kb]);
        acc1 = __builtin_amdgcn_wmma_f32_16x16x32_bf16(
            false, af, false, bfr, (short)0, acc1, false, false);
      }
    }
    __syncthreads();
    buf ^= 1;
  }

  // ---- store: C/D layout VGPR i -> M = i + 8*(lane>=16), N = lane&15 ----
  const int crow = m0 + wm * 32 + (lane >> 4) * 8;
  const int ccol = n0 + wn * 16 + (lane & 15);
  if (ccol < N) {
    const float bv = bias ? bias[ccol] : 0.0f;
    #pragma unroll
    for (int i = 0; i < 8; ++i) {
      C[(size_t)(crow + i)      * ldc + ccol] = acc0[i] + bv;
      C[(size_t)(crow + 16 + i) * ldc + ccol] = acc1[i] + bv;
    }
  }
}

// ---------------------------------------------------------------------------
// Grid-wide sense barrier (bar[0]=count, bar[1]=generation)
// ---------------------------------------------------------------------------
__device__ inline void grid_sync(int* bar, int nblocks) {
  __syncthreads();
  if (threadIdx.x == 0) {
    __threadfence();
    volatile int* gen = bar + 1;
    const int g = *gen;
    if (atomicAdd(bar, 1) == nblocks - 1) {
      *bar = 0;
      __threadfence();
      *gen = g + 1;
    } else {
      while (*gen == g) __builtin_amdgcn_s_sleep(1);
    }
    __threadfence();
  }
  __syncthreads();
}

// ---------------------------------------------------------------------------
// Persistent GRU. 128 blocks x 192 threads. Block b owns hidden units
// j0=8b..8b+7 and keeps the 24 matching W_hh rows (r,z,n) in LDS as bf16.
// Per step: stage h (4KB), 24 GEMV rows (8 threads/row, shfl reduce),
// gates + h_new for 8 units, one grid barrier.
// ---------------------------------------------------------------------------
__global__ __launch_bounds__(192)
void k_gru(const float* __restrict__ gi,    // [T, 3H] precomputed x@W_ih.T + b_ih
           const float* __restrict__ Whh,   // [3H, H] f32
           const float* __restrict__ bhh,   // [3H]
           float* __restrict__ hout,        // [T, H]
           float* __restrict__ hbuf,        // [H] working h
           int T, int* __restrict__ bar)
{
  __shared__ bf16_t sW[24 * HID];   // 48 KB
  __shared__ float  sh[HID];        // 4 KB
  __shared__ float  sgh[24];

  const int tid = threadIdx.x;
  const int j0  = blockIdx.x * 8;

  // one-time stage of this block's W_hh rows, f32 -> bf16
  for (int idx = tid; idx < 24 * HID; idx += 192) {
    const int r = idx >> 10, c = idx & (HID - 1);
    const int gate = r >> 3, jr = r & 7;
    sW[idx] = (bf16_t)Whh[(size_t)(gate * HID + j0 + jr) * HID + c];
  }

  const int row = tid >> 3;   // 0..23
  const int sub = tid & 7;    // 0..7  (128-wide K slice each)
  const float bias = bhh[(row >> 3) * HID + j0 + (row & 7)];

  for (int t = 0; t < T; ++t) {
    for (int i = tid; i < HID; i += 192) sh[i] = (t == 0) ? 0.0f : hbuf[i];
    __syncthreads();

    float s = 0.0f;
    const bf16_t* wr = &sW[row * HID + sub * 128];
    const float*  hr = &sh[sub * 128];
    #pragma unroll 8
    for (int k = 0; k < 128; ++k) s += (float)wr[k] * hr[k];
    s += __shfl_xor(s, 1);
    s += __shfl_xor(s, 2);
    s += __shfl_xor(s, 4);
    if (sub == 0) sgh[row] = s + bias;
    __syncthreads();

    if (tid < 8) {
      const int j = j0 + tid;
      const float* gir = gi + (size_t)t * H3;
      const float r_ = 1.0f / (1.0f + __expf(-(gir[j]           + sgh[tid])));
      const float z  = 1.0f / (1.0f + __expf(-(gir[HID + j]     + sgh[8 + tid])));
      const float n  = tanhf(gir[2 * HID + j] + r_ * sgh[16 + tid]);
      const float hn = (1.0f - z) * n + z * sh[j];
      hbuf[j] = hn;
      hout[(size_t)t * HID + j] = hn;
    }
    grid_sync(bar, gridDim.x);
  }
}

// ---------------------------------------------------------------------------
// Small elementwise / reduction kernels
// ---------------------------------------------------------------------------
__global__ void k_embed(const int* __restrict__ loc, const int* __restrict__ tim,
                        const int* __restrict__ uid,
                        const float* __restrict__ eloc, const float* __restrict__ etim,
                        const float* __restrict__ eusr, bf16_t* __restrict__ xb)
{
  const int t = blockIdx.x;
  const int l = loc[t], tm = tim[t], u = uid[0];
  for (int c = threadIdx.x; c < INP; c += blockDim.x) {
    float v;
    if (c < 1024)      v = eloc[(size_t)l * 1024 + c];
    else if (c < 1088) v = etim[(size_t)tm * 64 + (c - 1024)];
    else               v = eusr[(size_t)u * 64 + (c - 1088)];
    xb[(size_t)t * INP + c] = (bf16_t)v;
  }
}

__global__ void k_cvt_bf16(const float* __restrict__ s, bf16_t* __restrict__ d, size_t n) {
  for (size_t i = (size_t)blockIdx.x * blockDim.x + threadIdx.x; i < n;
       i += (size_t)gridDim.x * blockDim.x)
    d[i] = (bf16_t)s[i];
}

__global__ void k_transpose_cvt(const float* __restrict__ s, bf16_t* __restrict__ d,
                                int R, int Cc) {
  const size_t n = (size_t)R * Cc;
  for (size_t i = (size_t)blockIdx.x * blockDim.x + threadIdx.x; i < n;
       i += (size_t)gridDim.x * blockDim.x) {
    const int r = (int)(i / Cc), c = (int)(i % Cc);
    d[(size_t)c * R + r] = (bf16_t)s[i];
  }
}

__global__ void k_concat(const float* __restrict__ hs, const float* __restrict__ ctx,
                         bf16_t* __restrict__ outb) {
  const size_t n = (size_t)TGT * 2 * HID;
  for (size_t i = (size_t)blockIdx.x * blockDim.x + threadIdx.x; i < n;
       i += (size_t)gridDim.x * blockDim.x) {
    const int m = (int)(i >> 11), c = (int)(i & 2047);
    const float v = (c < HID) ? hs[(size_t)m * HID + c] : ctx[(size_t)m * HID + (c - HID)];
    outb[i] = (bf16_t)v;
  }
}

// row softmax of scores[1024,3072] -> bf16 probs
__global__ __launch_bounds__(256)
void k_softmax_rows(const float* __restrict__ S, bf16_t* __restrict__ P) {
  __shared__ float red[8];
  const int row = blockIdx.x, tid = threadIdx.x;
  const float* sr = S + (size_t)row * H3;
  float v[12];
  float mx = -INFINITY;
  #pragma unroll
  for (int i = 0; i < 12; ++i) { v[i] = sr[tid + i * 256]; mx = fmaxf(mx, v[i]); }
  for (int o = 16; o; o >>= 1) mx = fmaxf(mx, __shfl_xor(mx, o));
  if ((tid & 31) == 0) red[tid >> 5] = mx;
  __syncthreads();
  mx = red[0];
  #pragma unroll
  for (int i = 1; i < 8; ++i) mx = fmaxf(mx, red[i]);
  __syncthreads();
  float sum = 0.0f;
  #pragma unroll
  for (int i = 0; i < 12; ++i) { v[i] = __expf(v[i] - mx); sum += v[i]; }
  for (int o = 16; o; o >>= 1) sum += __shfl_xor(sum, o);
  if ((tid & 31) == 0) red[tid >> 5] = sum;
  __syncthreads();
  sum = 0.0f;
  #pragma unroll
  for (int i = 0; i < 8; ++i) sum += red[i];
  const float inv = 1.0f / sum;
  #pragma unroll
  for (int i = 0; i < 12; ++i) P[(size_t)row * H3 + tid + i * 256] = (bf16_t)(v[i] * inv);
}

// per-row streaming log-sum-exp over [1024, 50000]
__global__ __launch_bounds__(256)
void k_lse(const float* __restrict__ Y, float* __restrict__ stats) {
  __shared__ float smx[8], ssm[8];
  const int row = blockIdx.x, tid = threadIdx.x;
  const float* yr = Y + (size_t)row * NLOC;
  float mx = -INFINITY, s = 0.0f;
  for (int i = tid; i < NLOC; i += 256) {
    const float vv = yr[i];
    const float nm = fmaxf(mx, vv);
    s = s * __expf(mx - nm) + __expf(vv - nm);
    mx = nm;
  }
  for (int o = 16; o; o >>= 1) {
    const float om = __shfl_xor(mx, o), os = __shfl_xor(s, o);
    const float nm = fmaxf(mx, om);
    s = s * __expf(mx - nm) + os * __expf(om - nm);
    mx = nm;
  }
  if ((tid & 31) == 0) { smx[tid >> 5] = mx; ssm[tid >> 5] = s; }
  __syncthreads();
  if (tid == 0) {
    mx = smx[0]; s = ssm[0];
    #pragma unroll
    for (int i = 1; i < 8; ++i) {
      const float nm = fmaxf(mx, smx[i]);
      s = s * __expf(mx - nm) + ssm[i] * __expf(smx[i] - nm);
      mx = nm;
    }
    stats[row] = mx + __logf(s);
  }
}

__global__ void k_sub_stats(float* __restrict__ Y, const float* __restrict__ stats) {
  const size_t n = (size_t)TGT * NLOC;
  for (size_t i = (size_t)blockIdx.x * blockDim.x + threadIdx.x; i < n;
       i += (size_t)gridDim.x * blockDim.x)
    Y[i] -= stats[i / NLOC];
}

// ---------------------------------------------------------------------------
// Host launcher
// ---------------------------------------------------------------------------
extern "C" void kernel_launch(void* const* d_in, const int* in_sizes, int n_in,
                              void* d_out, int out_size, void* d_ws, size_t ws_size,
                              hipStream_t stream) {
  const int*   loc      = (const int*)d_in[0];
  const int*   tim      = (const int*)d_in[1];
  const int*   uid      = (const int*)d_in[2];
  const float* emb_loc  = (const float*)d_in[4];
  const float* emb_tim  = (const float*)d_in[5];
  const float* emb_user = (const float*)d_in[6];
  const float* enc_W_ih = (const float*)d_in[7];
  const float* enc_W_hh = (const float*)d_in[8];
  const float* enc_b_ih = (const float*)d_in[9];
  const float* enc_b_hh = (const float*)d_in[10];
  const float* dec_W_ih = (const float*)d_in[11];
  const float* dec_W_hh = (const float*)d_in[12];
  const float* dec_b_ih = (const float*)d_in[13];
  const float* dec_b_hh = (const float*)d_in[14];
  const float* fc_W     = (const float*)d_in[15];
  const float* fc_b     = (const float*)d_in[16];
  float* out = (float*)d_out;

  // workspace carve-out (~119 MB total)
  char* base = (char*)d_ws;
  size_t off = 0;
  auto carve = [&](size_t bytes) { char* p = base + off; off = (off + bytes + 255) & ~(size_t)255; return p; };
  bf16_t* xb      = (bf16_t*)carve((size_t)SEQ * INP * 2);
  float*  gi      = (float*) carve((size_t)SEQ * H3 * 4);
  float*  hh      = (float*) carve((size_t)ENC_T * HID * 4);
  float*  hs      = (float*) carve((size_t)TGT * HID * 4);
  float*  hbuf    = (float*) carve((size_t)HID * 4);
  bf16_t* hh_b    = (bf16_t*)carve((size_t)ENC_T * HID * 2);
  bf16_t* hhT_b   = (bf16_t*)carve((size_t)HID * ENC_T * 2);
  bf16_t* hs_b    = (bf16_t*)carve((size_t)TGT * HID * 2);
  float*  scores  = (float*) carve((size_t)TGT * ENC_T * 4);
  bf16_t* P_b     = (bf16_t*)carve((size_t)TGT * ENC_T * 2);
  float*  ctx     = (float*) carve((size_t)TGT * HID * 4);
  bf16_t* outb    = (bf16_t*)carve((size_t)TGT * 2 * HID * 2);
  float*  stats   = (float*) carve((size_t)TGT * 4);
  int*    bar     = (int*)   carve(256);
  (void)ws_size; (void)in_sizes; (void)n_in; (void)out_size;

  // 1) embeddings -> x (bf16)
  k_embed<<<SEQ, 256, 0, stream>>>(loc, tim, uid, emb_loc, emb_tim, emb_user, xb);

  // 2) batched input GEMMs: gi = x @ W_ih^T + b_ih   (encoder rows, decoder rows)
  k_gemm_nt<true><<<dim3(ENC_T / 64, H3 / 64), 256, 0, stream>>>(
      xb, enc_W_ih, enc_b_ih, gi, ENC_T, H3, INP, INP, INP, H3);
  k_gemm_nt<true><<<dim3(TGT / 64, H3 / 64), 256, 0, stream>>>(
      xb + (size_t)ENC_T * INP, dec_W_ih, dec_b_ih, gi + (size_t)ENC_T * H3,
      TGT, H3, INP, INP, INP, H3);

  // 3) persistent GRUs (grid barrier state must start at 0 each call)
  hipMemsetAsync(bar, 0, 64, stream);
  k_gru<<<HID / 8, 192, 0, stream>>>(gi, enc_W_hh, enc_b_hh, hh, hbuf, ENC_T, bar);
  k_gru<<<HID / 8, 192, 0, stream>>>(gi + (size_t)ENC_T * H3, dec_W_hh, dec_b_hh,
                                     hs, hbuf, TGT, bar + 8);

  // 4) bf16 copies / transpose for attention GEMMs
  k_cvt_bf16<<<2048, 256, 0, stream>>>(hh, hh_b, (size_t)ENC_T * HID);
  k_cvt_bf16<<<1024, 256, 0, stream>>>(hs, hs_b, (size_t)TGT * HID);
  k_transpose_cvt<<<2048, 256, 0, stream>>>(hh, hhT_b, ENC_T, HID);

  // 5) attention: scores = hs @ hh^T ; softmax ; ctx = P @ hh
  k_gemm_nt<false><<<dim3(TGT / 64, ENC_T / 64), 256, 0, stream>>>(
      hs_b, hh_b, nullptr, scores, TGT, ENC_T, HID, HID, HID, ENC_T);
  k_softmax_rows<<<TGT, 256, 0, stream>>>(scores, P_b);
  k_gemm_nt<false><<<dim3(TGT / 64, HID / 64), 256, 0, stream>>>(
      P_b, hhT_b, nullptr, ctx, TGT, HID, ENC_T, ENC_T, ENC_T, HID);

  // 6) concat -> bf16, FC GEMM straight into d_out (fc_W converted on the fly)
  k_concat<<<2048, 256, 0, stream>>>(hs, ctx, outb);
  k_gemm_nt<true><<<dim3(TGT / 64, (NLOC + 63) / 64), 256, 0, stream>>>(
      outb, fc_W, fc_b, out, TGT, NLOC, 2 * HID, 2 * HID, 2 * HID, NLOC);

  // 7) in-place log-softmax
  k_lse<<<TGT, 256, 0, stream>>>(out, stats);
  k_sub_stats<<<4096, 256, 0, stream>>>(out, stats);
}
